// GCN_69466801045656
// MI455X (gfx1250) — matrix operand order repaired and checked by reading, not compiled, not run on hardware
//
#include <hip/hip_runtime.h>
#include <hip/hip_bf16.h>

#define N_NODES 50000   // multiple of 16: GEMM tiles need no row guards

typedef __attribute__((ext_vector_type(16))) __bf16 v16bf;
typedef __attribute__((ext_vector_type(8)))  __bf16 v8bf;
typedef __attribute__((ext_vector_type(4)))  __bf16 v4bf;
typedef __attribute__((ext_vector_type(8)))  float  v8f;

// ---------------- one-time weight conversion to bf16 ----------------
__global__ void f32_to_bf16_kernel(const float* __restrict__ in,
                                   __bf16* __restrict__ out, int n) {
    int i = blockIdx.x * blockDim.x + threadIdx.x;
    if (i < n) out[i] = (__bf16)in[i];
}

// ---------------- degree / normalization ----------------
__global__ void deg_init_kernel(float* deg, int n) {
    int i = blockIdx.x * blockDim.x + threadIdx.x;
    if (i < n) deg[i] = 1.0f;                 // self-loop
}

__global__ void deg_count_kernel(const long long* __restrict__ dst, float* deg, int e) {
    int i = blockIdx.x * blockDim.x + threadIdx.x;
    if (i < e) atomicAdd(deg + dst[i], 1.0f);
}

__global__ void deg_rsqrt_kernel(float* deg, int n) {
    int i = blockIdx.x * blockDim.x + threadIdx.x;
    if (i < n) deg[i] = rsqrtf(deg[i]);
}

// ---------------- per-edge prep: int32 indices + edge norm, computed once ---------
__global__ void edge_prep_kernel(const long long* __restrict__ src,
                                 const long long* __restrict__ dst,
                                 const float* __restrict__ dinv,
                                 int* __restrict__ srcI, int* __restrict__ dstI,
                                 float* __restrict__ norm, int e) {
    int i = blockIdx.x * blockDim.x + threadIdx.x;
    if (i >= e) return;
    long long s = src[i], d = dst[i];
    srcI[i] = (int)s;
    dstI[i] = (int)d;
    norm[i] = dinv[s] * dinv[d];
}

// ---------------- WMMA GEMM: H[M,OUT] = A[M,KDIM] @ Wb[KDIM,OUT] (bf16 weights) ----
// One block = 16-row strip of A (requires M % 16 == 0), staged in LDS as bf16.
// One wave per 16-col tile; blockDim.x must be OUT*2.
template<int KDIM, int OUT, bool RELU>
__global__ void gemm_wmma_kernel(const float* __restrict__ A,
                                 const __bf16* __restrict__ Wb,
                                 float* __restrict__ H) {
    constexpr int LROW = KDIM + 8;            // +8 bf16 (16B) pad: spreads banks
    __shared__ __align__(16) __bf16 xs[16 * LROW];
    const int tid      = threadIdx.x;
    const int row_base = blockIdx.x * 16;

    // stage A tile: float4 loads -> bf16 convert once -> 8B LDS stores
    for (int i = tid * 4; i < 16 * KDIM; i += blockDim.x * 4) {
        int r = i / KDIM, k = i - r * KDIM;   // k is a multiple of 4
        float4 v = *(const float4*)(A + (size_t)(row_base + r) * KDIM + k);
        if (RELU) {
            v.x = fmaxf(v.x, 0.f); v.y = fmaxf(v.y, 0.f);
            v.z = fmaxf(v.z, 0.f); v.w = fmaxf(v.w, 0.f);
        }
        v4bf b4 = { (__bf16)v.x, (__bf16)v.y, (__bf16)v.z, (__bf16)v.w };
        *(v4bf*)(xs + r * LROW + k) = b4;
    }
    __syncthreads();

    const int lane   = tid & 31;
    const int wave   = tid >> 5;
    const int n_base = wave * 16;
    const int m      = lane & 15;             // A row / D column select
    const int kh     = lane >> 4;             // K-half (A) & M-half (D)

    v8f c = {};
    #pragma unroll
    for (int kk = 0; kk < KDIM; kk += 32) {
        // A fragment (ISA 7.12.2, 16-bit A 16x32): lane L -> M=L&15;
        // elems 0-7: K = kk + kh*8 + e ; elems 8-15: K = kk + 16 + kh*8 + (e-8)
        const v8bf a_lo = *(const v8bf*)(xs + m * LROW + kk + kh * 8);
        const v8bf a_hi = *(const v8bf*)(xs + m * LROW + kk + 16 + kh * 8);
        v16bf a = __builtin_shufflevector(a_lo, a_hi,
                    0,1,2,3,4,5,6,7,8,9,10,11,12,13,14,15);

        // B fragment (32x16): lane L -> N = n_base+(L&15); elems 0-7: K = kk + kh*8 + e ; elems 8-15: K = kk + 16 + kh*8 + (e-8)
        const __bf16* wrow = Wb + (size_t)(kk + kh * 8) * OUT + n_base + m;
        v16bf b;
        #pragma unroll
        for (int e = 0; e < 8; ++e) {
            b[e]     = wrow[(size_t)e * OUT];
            b[e + 8] = wrow[(size_t)(16 + e) * OUT];
        }

        c = __builtin_amdgcn_wmma_f32_16x16x32_bf16(
                false, a, false, b, (short)0, c, false, false);
    }

    // D layout: VGPR r -> M = r + kh*8, N = lane&15. Full tiles -> no guards:
    // single base address, 8 stores at immediate offsets r*OUT*4.
    float* hbase = H + (size_t)(row_base + kh * 8) * OUT + n_base + (lane & 15);
    #pragma unroll
    for (int r = 0; r < 8; ++r) hbase[(size_t)r * OUT] = c[r];
}

// ---------------- aggregation init: out = h*dinv^2 + bias (4 features/thread) ------
template<int F>
__global__ void agg_init_kernel(const float* __restrict__ h,
                                const float* __restrict__ dinv,
                                const float* __restrict__ bias,
                                float* __restrict__ out, long long total4) {
    long long idx = (long long)blockIdx.x * blockDim.x + threadIdx.x;
    if (idx >= total4) return;
    constexpr int F4 = F / 4;
    int i = (int)(idx / F4);
    int f = (int)(idx - (long long)i * F4) * 4;
    float d  = dinv[i];
    float d2 = d * d;
    float4 v = *(const float4*)(h + (size_t)i * F + f);
    float4 b = *(const float4*)(bias + f);
    float4 o = make_float4(v.x * d2 + b.x, v.y * d2 + b.y,
                           v.z * d2 + b.z, v.w * d2 + b.w);
    *(float4*)(out + (size_t)i * F + f) = o;
}

// ---------------- edge scatter: out[dst] += h[src]*norm (4 features/thread) --------
template<int F>
__global__ void agg_scatter_kernel(const float* __restrict__ h,
                                   const float* __restrict__ norm,
                                   const int* __restrict__ srcI,
                                   const int* __restrict__ dstI,
                                   float* out, long long total4) {
    long long idx = (long long)blockIdx.x * blockDim.x + threadIdx.x;
    if (idx >= total4) return;
    constexpr int F4 = F / 4;
    long long e = idx / F4;
    int f = (int)(idx - e * F4) * 4;          // consecutive lanes -> consecutive feats
    int s = srcI[e], d = dstI[e];
    float nrm = norm[e];
    float4 v = *(const float4*)(h + (size_t)s * F + f);
    float* o = out + (size_t)d * F + f;
    atomicAdd(o + 0, v.x * nrm);
    atomicAdd(o + 1, v.y * nrm);
    atomicAdd(o + 2, v.z * nrm);
    atomicAdd(o + 3, v.w * nrm);
}

// ---------------- log_softmax over 64-wide rows, one wave32 per row ----------------
__global__ void logsoftmax64_kernel(float* out, int nrows) {
    int row = blockIdx.x * (blockDim.x >> 5) + (threadIdx.x >> 5);
    if (row >= nrows) return;                 // uniform per wave
    int lane = threadIdx.x & 31;
    float* p = out + (size_t)row * 64;
    float v0 = p[lane], v1 = p[lane + 32];
    float m = fmaxf(v0, v1);
    #pragma unroll
    for (int off = 16; off; off >>= 1) m = fmaxf(m, __shfl_xor(m, off, 32));
    float s = expf(v0 - m) + expf(v1 - m);
    #pragma unroll
    for (int off = 16; off; off >>= 1) s += __shfl_xor(s, off, 32);
    float l = logf(s);
    p[lane]      = v0 - m - l;
    p[lane + 32] = v1 - m - l;
}

// ---------------- driver ----------------
extern "C" void kernel_launch(void* const* d_in, const int* in_sizes, int n_in,
                              void* d_out, int out_size, void* d_ws, size_t ws_size,
                              hipStream_t stream) {
    const float*     x  = (const float*)d_in[0];
    const long long* ei = (const long long*)d_in[1];   // int64 [2, E]
    const float*     W1 = (const float*)d_in[2];
    const float*     b1 = (const float*)d_in[3];
    const float*     W2 = (const float*)d_in[4];
    const float*     b2 = (const float*)d_in[5];

    const int N = N_NODES;
    const int E = in_sizes[1] / 2;                     // 800000
    const long long* src = ei;
    const long long* dst = ei + E;

    // workspace layout (all offsets 256B-aligned; E assumed <= 800000)
    char* ws = (char*)d_ws;
    float*  dinv = (float*) (ws + 0);                  // N f32 (deg -> rsqrt)
    float*  h1   = (float*) (ws + 200192);             // N*128 f32
    float*  a1   = (float*) (ws + 25800192);           // N*128 f32
    float*  h2   = (float*) (ws + 51400192);           // N*64  f32
    __bf16* W1b  = (__bf16*)(ws + 64200192);           // 128*128 bf16
    __bf16* W2b  = (__bf16*)(ws + 64232960);           // 128*64  bf16
    int*    srcI = (int*)   (ws + 64249344);           // E i32
    int*    dstI = (int*)   (ws + 67449344);           // E i32
    float*  norm = (float*) (ws + 70649344);           // E f32
    float*  a2   = (float*)d_out;                      // N*64 final buffer

    // 0) one-time weight conversion to bf16
    f32_to_bf16_kernel<<<(128 * 128 + 255) / 256, 256, 0, stream>>>(W1, W1b, 128 * 128);
    f32_to_bf16_kernel<<<(128 * 64 + 255) / 256, 256, 0, stream>>>(W2, W2b, 128 * 64);

    // 1) degrees -> dinv -> per-edge norms + int32 indices
    deg_init_kernel <<<(N + 255) / 256, 256, 0, stream>>>(dinv, N);
    deg_count_kernel<<<(E + 255) / 256, 256, 0, stream>>>(dst, dinv, E);
    deg_rsqrt_kernel<<<(N + 255) / 256, 256, 0, stream>>>(dinv, N);
    edge_prep_kernel<<<(E + 255) / 256, 256, 0, stream>>>(src, dst, dinv,
                                                          srcI, dstI, norm, E);

    const int mblocks = N / 16;                        // 3125, exact

    // 2) layer 1: h1 = x @ W1  (bf16 WMMA, f32 accum)
    gemm_wmma_kernel<128, 128, false><<<mblocks, 256, 0, stream>>>(x, W1b, h1);

    // 3) layer-1 aggregation (self-loop + bias, then edge scatter)
    {
        long long tn4 = (long long)N * 32;             // N*128/4
        long long te4 = (long long)E * 32;             // E*128/4
        agg_init_kernel<128>   <<<(unsigned)((tn4 + 255) / 256), 256, 0, stream>>>(h1, dinv, b1, a1, tn4);
        agg_scatter_kernel<128><<<(unsigned)((te4 + 255) / 256), 256, 0, stream>>>(h1, norm, srcI, dstI, a1, te4);
    }

    // 4) layer 2: h2 = relu(a1) @ W2   (ReLU fused into LDS A-tile staging)
    gemm_wmma_kernel<128, 64, true><<<mblocks, 128, 0, stream>>>(a1, W2b, h2);

    // 5) layer-2 aggregation directly into d_out
    {
        long long tn4 = (long long)N * 16;             // N*64/4
        long long te4 = (long long)E * 16;             // E*64/4
        agg_init_kernel<64>   <<<(unsigned)((tn4 + 255) / 256), 256, 0, stream>>>(h2, dinv, b2, a2, tn4);
        agg_scatter_kernel<64><<<(unsigned)((te4 + 255) / 256), 256, 0, stream>>>(h2, norm, srcI, dstI, a2, te4);
    }

    // 6) log_softmax in place on d_out
    logsoftmax64_kernel<<<(N * 32 + 255) / 256, 256, 0, stream>>>(a2, N);
}
